// MirrorResonance_5007931867587
// MI455X (gfx1250) — matrix operand order repaired and checked by reading, not compiled, not run on hardware
//
#include <hip/hip_runtime.h>

typedef __attribute__((ext_vector_type(8)))  __bf16 bhalf8;
typedef __attribute__((ext_vector_type(16))) __bf16 bhalf16;
typedef __attribute__((ext_vector_type(8)))  float  vfloat8;

#define DIM 1024
#define NA  8
#define INV2PI 0.15915494309189535f

__device__ __forceinline__ unsigned short f2bf(float f) {
  unsigned int u = __float_as_uint(f);
  u += 0x7FFFu + ((u >> 16) & 1u);           // round-to-nearest-even
  return (unsigned short)(u >> 16);
}

__device__ __forceinline__ float fast_tanh(float x) {
  float e = __expf(x + x);
  return 1.0f - 2.0f / (e + 1.0f);
}

// ---------------------------------------------------------------- kernel 1:
// X (T,1024) f32 -> bf16, vectorized 4-wide
__global__ __launch_bounds__(256) void cvt_bf16_kernel(
    const float* __restrict__ in, unsigned int* __restrict__ out, int n4) {
  int i = blockIdx.x * 256 + threadIdx.x;
  if (i < n4) {
    float4 v = ((const float4*)in)[i];
    unsigned int p0 = (unsigned int)f2bf(v.x) | ((unsigned int)f2bf(v.y) << 16);
    unsigned int p1 = (unsigned int)f2bf(v.z) | ((unsigned int)f2bf(v.w) << 16);
    ((uint2*)out)[i] = make_uint2(p0, p1);
  }
}

// ---------------------------------------------------------------- kernel 2:
// W1 (D,D) f32 row-major -> W1 bf16 (n-major): w1t[n*D + k] = bf16(W1[n*D + k])
__global__ __launch_bounds__(256) void w1t_kernel(
    const float* __restrict__ W1, unsigned short* __restrict__ w1t) {
  __shared__ unsigned short tile[32][33];
  const int tx = threadIdx.x, ty = threadIdx.y;   // (32, 8)
  const int k0 = blockIdx.x * 32, n0 = blockIdx.y * 32;
  #pragma unroll
  for (int jj = 0; jj < 4; ++jj)
    tile[ty + jj * 8][tx] = f2bf(W1[(size_t)(n0 + ty + jj * 8) * DIM + k0 + tx]);
  __syncthreads();
  #pragma unroll
  for (int jj = 0; jj < 4; ++jj)
    w1t[(size_t)(n0 + ty + jj * 8) * DIM + k0 + tx] = tile[ty + jj * 8][tx];
}

// ---------------------------------------------------------------- kernel 3:
// Fused encoder: per wave, two 16-row blocks of E' = (tanh(X@W1^T + b1) @ W2^T + b2) / 2pi
// A fragment (16x32 bf16): lane 0-15 -> M=lane, K {0..7,16..23}; lane 16-31 -> M=lane-16, K {8..15,24..31}
// B fragment (32x16 bf16): lane -> N=lane%16, K same as A fragment  => two 8-wide bf16 runs of W1 row n
__global__ __launch_bounds__(128) void encoder_kernel(
    const unsigned short* __restrict__ xb,   // (T, D) bf16
    const unsigned short* __restrict__ w1t,  // (D, D) bf16, [n][k]
    const float* __restrict__ b1,            // (D)
    const float* __restrict__ W2,            // (A, D)
    const float* __restrict__ b2,            // (A)
    float* __restrict__ ep)                  // (T, A)  E/2pi
{
  const int lane = threadIdx.x & 31;
  const int wid  = blockIdx.x * 4 + (threadIdx.x >> 5);  // wave id: 32 rows each
  const int mrow = lane & 15;
  const int koff = (lane >> 4) * 8;

  const unsigned short* a0p = xb + (size_t)(wid * 32 + mrow) * DIM + koff;
  const unsigned short* a1p = a0p + (size_t)16 * DIM;
  const unsigned short* bp  = w1t + (size_t)mrow * DIM + koff;

  float e0[8][8], e1[8][8];
  #pragma unroll
  for (int r = 0; r < 8; ++r)
    #pragma unroll
    for (int a = 0; a < 8; ++a) { e0[r][a] = 0.f; e1[r][a] = 0.f; }

  for (int j = 0; j < DIM / 16; ++j) {
    vfloat8 c0 = {0.f, 0.f, 0.f, 0.f, 0.f, 0.f, 0.f, 0.f};
    vfloat8 c1 = {0.f, 0.f, 0.f, 0.f, 0.f, 0.f, 0.f, 0.f};
    const unsigned short* bj = bp + (size_t)j * 16 * DIM;
    #pragma unroll 2
    for (int kc = 0; kc < DIM; kc += 32) {
      bhalf8 bl = *(const bhalf8*)(bj + kc);
      bhalf8 bh = *(const bhalf8*)(bj + kc + 16);
      bhalf16 B = __builtin_shufflevector(bl, bh, 0,1,2,3,4,5,6,7,8,9,10,11,12,13,14,15);
      bhalf8 al0 = *(const bhalf8*)(a0p + kc);
      bhalf8 ah0 = *(const bhalf8*)(a0p + kc + 16);
      bhalf16 A0 = __builtin_shufflevector(al0, ah0, 0,1,2,3,4,5,6,7,8,9,10,11,12,13,14,15);
      bhalf8 al1 = *(const bhalf8*)(a1p + kc);
      bhalf8 ah1 = *(const bhalf8*)(a1p + kc + 16);
      bhalf16 A1 = __builtin_shufflevector(al1, ah1, 0,1,2,3,4,5,6,7,8,9,10,11,12,13,14,15);
      c0 = __builtin_amdgcn_wmma_f32_16x16x32_bf16(false, A0, false, B, (short)0, c0, false, false);
      c1 = __builtin_amdgcn_wmma_f32_16x16x32_bf16(false, A1, false, B, (short)0, c1, false, false);
    }
    const int ng = j * 16 + mrow;
    const float bn = b1[ng];
    float w2v[8];
    #pragma unroll
    for (int a = 0; a < 8; ++a) w2v[a] = W2[a * DIM + ng];
    #pragma unroll
    for (int r = 0; r < 8; ++r) {
      float h0 = fast_tanh(c0[r] + bn);
      float h1 = fast_tanh(c1[r] + bn);
      #pragma unroll
      for (int a = 0; a < 8; ++a) {
        e0[r][a] = __builtin_fmaf(h0, w2v[a], e0[r][a]);
        e1[r][a] = __builtin_fmaf(h1, w2v[a], e1[r][a]);
      }
    }
  }

  // reduce over the 16 N-lanes of each half (masks < 16 stay within halves)
  #pragma unroll
  for (int mask = 1; mask <= 8; mask <<= 1)
    #pragma unroll
    for (int r = 0; r < 8; ++r)
      #pragma unroll
      for (int a = 0; a < 8; ++a) {
        e0[r][a] += __shfl_xor(e0[r][a], mask, 32);
        e1[r][a] += __shfl_xor(e1[r][a], mask, 32);
      }

  // lanes {a, 16+a} write column a; low half holds M=0..7, high half M=8..15
  const int hb = (lane >> 4) * 8;
  #pragma unroll
  for (int r = 0; r < 8; ++r)
    #pragma unroll
    for (int a = 0; a < 8; ++a)
      if ((lane & 7) == a && (lane & 15) < 8) {
        int t0 = (wid * 2) * 16 + r + hb;
        int t1 = (wid * 2 + 1) * 16 + r + hb;
        ep[t0 * NA + a] = (e0[r][a] + b2[a]) * INV2PI;
        ep[t1 * NA + a] = (e1[r][a] + b2[a]) * INV2PI;
      }
}

// ---------------------------------------------------------------- kernel 4:
// Sequential Kuramoto entrainment, rotation units. q_{t+1} = q + f*dt + (K/2pi)*sin2pi(e_t - q)
__global__ __launch_bounds__(32) void scan_kernel(
    const float* __restrict__ ep, const float* __restrict__ freqs,
    const float* __restrict__ ph0, float* __restrict__ qout, int T) {
  const int a = threadIdx.x;
  if (a >= NA) return;
  const float bstep = freqs[a] * 0.01f;      // base/2pi = freq*dt
  const float kp    = 0.5f * INV2PI;         // K/2pi
  float q = ph0[a] * INV2PI;

  float buf0[16], buf1[16];
  #pragma unroll
  for (int i = 0; i < 16; ++i) buf0[i] = ep[i * NA + a];
  const int nch = T / 16;
  for (int ch = 0; ch < nch; ch += 2) {
    #pragma unroll
    for (int i = 0; i < 16; ++i) buf1[i] = ep[((ch + 1) * 16 + i) * NA + a];
    #pragma unroll
    for (int i = 0; i < 16; ++i) {
      float s = __builtin_amdgcn_sinf(buf0[i] - q);   // v_sin_f32: rotations in
      q = __builtin_fmaf(kp, s, q + bstep);
    }
    if (ch + 2 < nch) {
      #pragma unroll
      for (int i = 0; i < 16; ++i) buf0[i] = ep[((ch + 2) * 16 + i) * NA + a];
    }
    #pragma unroll
    for (int i = 0; i < 16; ++i) {
      float s = __builtin_amdgcn_sinf(buf1[i] - q);
      q = __builtin_fmaf(kp, s, q + bstep);
    }
    q -= floorf(q);                                   // wrap every 32 steps
  }
  q -= floorf(q);
  qout[a] = q;
}

// ---------------------------------------------------------------- kernel 5:
// reproduced[s,d] = dec_b[d] + sum_a cos2pi(q[a] + (s+1)*freq[a]*dt) * dec_W[d,a]
__global__ __launch_bounds__(256) void repro_kernel(
    const float* __restrict__ q, const float* __restrict__ freqs,
    const float* __restrict__ decW, const float* __restrict__ decb,
    float* __restrict__ out) {
  __shared__ float cs[NA];
  const int s = blockIdx.x;
  const float t = (float)(s + 1);
  if (threadIdx.x < NA) {
    int a = threadIdx.x;
    cs[a] = __builtin_amdgcn_cosf(q[a] + t * (freqs[a] * 0.01f));  // v_cos_f32
  }
  __syncthreads();
  for (int d = threadIdx.x; d < DIM; d += 256) {
    const float4* w = (const float4*)(decW + (size_t)d * NA);
    float4 w0 = w[0], w1 = w[1];
    float r = decb[d];
    r = __builtin_fmaf(cs[0], w0.x, r);
    r = __builtin_fmaf(cs[1], w0.y, r);
    r = __builtin_fmaf(cs[2], w0.z, r);
    r = __builtin_fmaf(cs[3], w0.w, r);
    r = __builtin_fmaf(cs[4], w1.x, r);
    r = __builtin_fmaf(cs[5], w1.y, r);
    r = __builtin_fmaf(cs[6], w1.z, r);
    r = __builtin_fmaf(cs[7], w1.w, r);
    out[(size_t)s * DIM + d] = r;
  }
}

// ----------------------------------------------------------------
extern "C" void kernel_launch(void* const* d_in, const int* in_sizes, int n_in,
                              void* d_out, int out_size, void* d_ws, size_t ws_size,
                              hipStream_t stream) {
  const float* X     = (const float*)d_in[0];
  const float* W1    = (const float*)d_in[1];
  const float* b1    = (const float*)d_in[2];
  const float* W2    = (const float*)d_in[3];
  const float* b2    = (const float*)d_in[4];
  const float* freqs = (const float*)d_in[5];
  const float* decW  = (const float*)d_in[6];
  const float* decb  = (const float*)d_in[7];
  const float* ph0   = (const float*)d_in[8];
  (void)n_in;

  const int T = in_sizes[0] / DIM;   // 16384
  const int S = out_size / DIM;      // num_steps (avoid reading device scalar)

  char* ws = (char*)d_ws;
  unsigned short* xb  = (unsigned short*)ws;                                  // T*D*2 B
  unsigned short* w1t = (unsigned short*)(ws + (size_t)T * DIM * 2);          // D*D*2 B
  float* epd = (float*)(ws + (size_t)T * DIM * 2 + (size_t)DIM * DIM * 2);    // T*A*4 B
  float* qd  = epd + (size_t)T * NA;                                          // A*4 B
  (void)ws_size;

  const int n4 = T * DIM / 4;
  cvt_bf16_kernel<<<n4 / 256, 256, 0, stream>>>(X, (unsigned int*)xb, n4);
  w1t_kernel<<<dim3(DIM / 32, DIM / 32), dim3(32, 8), 0, stream>>>(W1, w1t);
  encoder_kernel<<<T / 128, 128, 0, stream>>>(xb, w1t, b1, W2, b2, epd);
  scan_kernel<<<1, 32, 0, stream>>>(epd, freqs, ph0, qd, T);
  repro_kernel<<<S, 256, 0, stream>>>(qd, freqs, decW, decb, (float*)d_out);
}
